// AttentionLayer_1477468749950
// MI455X (gfx1250) — matrix operand order repaired and checked by reading, not compiled, not run on hardware
//
#include <hip/hip_runtime.h>
#include <hip/hip_bf16.h>

// ---------------------------------------------------------------------------
// Attention layer for MI455X (gfx1250), wave32 + WMMA f16 (f32 accumulate).
//   q = Wq x + bq ; k = Wk x + bk ; v = Wv x + bv   (1x1 convs = GEMMs)
//   out = v @ softmax(q^T k)^T + x
// Flash-style attention: 32 queries x 512 channels per workgroup (8 waves),
// K tiles staged in LDS via GLOBAL_LOAD_ASYNC_TO_LDS_B128, V fragments
// reused across two query tiles, f32 softmax with one shfl_xor(16) butterfly.
// ---------------------------------------------------------------------------

typedef __attribute__((ext_vector_type(16))) _Float16     v16h;
typedef __attribute__((ext_vector_type(8)))  float        v8f;
typedef __attribute__((ext_vector_type(4)))  float        v4f;
typedef __attribute__((ext_vector_type(4)))  unsigned int v4u;
typedef __attribute__((ext_vector_type(4)))  int          v4i;

typedef __attribute__((address_space(1))) v4i* gv4i_p;  // global <4 x i32>*
typedef __attribute__((address_space(3))) v4i* lv4i_p;  // LDS    <4 x i32>*

union H8 { v4u u; _Float16 h[8]; };

#define WMMA_F16(a, b, c) \
  __builtin_amdgcn_wmma_f32_16x16x32_f16(false, (a), false, (b), (short)0, (c), false, false)

#if defined(__has_builtin)
#if __has_builtin(__builtin_amdgcn_global_load_async_to_lds_b128) && \
    __has_builtin(__builtin_amdgcn_s_wait_asynccnt)
#define USE_ASYNC_LDS 1
#endif
#endif

// A-matrix (16x32 f16) fragment: lane (h = lane/16, m = lane%16) holds row m.
// elements 0..7  <- cols kofs + h*16 + {0..7}
// elements 8..15 <- cols kofs + h*16 + 8 + {0..7}
static __device__ inline v16h load_a16(const _Float16* row, int kofs, int h) {
  H8 lo, hi;
  lo.u = *(const v4u*)(row + kofs + h * 16);
  hi.u = *(const v4u*)(row + kofs + h * 16 + 8);
  v16h a;
#pragma unroll
  for (int t = 0; t < 8; ++t) { a[t] = lo.h[t]; a[8 + t] = hi.h[t]; }
  return a;
}

// B-matrix (32x16 f16) fragment: lane (h, n = lane%16) holds column n.
// element t <- row kofs + h*16 + t  (contiguous when source is K-contiguous).
static __device__ inline v16h load_b16(const _Float16* col, int kofs, int h) {
  H8 lo, hi;
  lo.u = *(const v4u*)(col + kofs + h * 16);
  hi.u = *(const v4u*)(col + kofs + h * 16 + 8);
  v16h b;
#pragma unroll
  for (int t = 0; t < 8; ++t) { b[t] = lo.h[t]; b[8 + t] = hi.h[t]; }
  return b;
}

// ---------------------------------------------------------------------------
// Projection GEMM: out[o,n] = sum_c W[o,c] * x[b,c,n] + bias[o], f16 output.
// Block = 256 (8 waves). Wave w computes a 64(o) x 16(n) strip: the strided
// x B-fragment is gathered once per k-step and reused by 4 WMMAs.
// transpose_out = 1 -> out stored [N, O] (q,k); 0 -> [O, N] (v).
// Grid: (N/128, O/64, B).
// ---------------------------------------------------------------------------
__global__ void __launch_bounds__(256)
proj64_kernel(const float* __restrict__ x, const float* __restrict__ W,
              const float* __restrict__ bias, _Float16* __restrict__ out,
              int O, int transpose_out) {
  const int N = 4096, Cc = 512;
  const int lane = threadIdx.x & 31;
  const int wave = threadIdx.x >> 5;
  const int h = lane >> 4, t16 = lane & 15;
  const int n0 = blockIdx.x * 128 + wave * 16;
  const int o0 = blockIdx.y * 64;
  const int b  = blockIdx.z;
  const float* xb = x + (size_t)b * Cc * N;
  out += (size_t)b * (size_t)O * N;

  v8f acc[4] = {{}, {}, {}, {}};
  for (int kk = 0; kk < Cc; kk += 32) {
    // B = x tile (rows c, cols n): column gather, stride N (L2 resident).
    const float* xcol = xb + (size_t)(kk + h * 16) * N + n0 + t16;
    v16h bm;
#pragma unroll
    for (int t = 0; t < 16; ++t) bm[t] = (_Float16)xcol[(size_t)t * N];
#pragma unroll
    for (int sub = 0; sub < 4; ++sub) {
      const float* wrow = W + (size_t)(o0 + sub * 16 + t16) * Cc + kk;
      v4f a0 = *(const v4f*)(wrow + h * 16);
      v4f a1 = *(const v4f*)(wrow + h * 16 + 4);
      v4f a2 = *(const v4f*)(wrow + h * 16 + 8);
      v4f a3 = *(const v4f*)(wrow + h * 16 + 12);
      v16h a;
#pragma unroll
      for (int t = 0; t < 4; ++t) {
        a[t]      = (_Float16)a0[t];
        a[4 + t]  = (_Float16)a1[t];
        a[8 + t]  = (_Float16)a2[t];
        a[12 + t] = (_Float16)a3[t];
      }
      acc[sub] = WMMA_F16(a, bm, acc[sub]);
    }
  }

  // D layout: VGPR r, lane -> row o = o0 + sub*16 + 8h + r, col n = n0 + t16.
#pragma unroll
  for (int sub = 0; sub < 4; ++sub) {
    const int ob = o0 + sub * 16 + 8 * h;
    if (transpose_out) {
      H8 pk;
#pragma unroll
      for (int r = 0; r < 8; ++r) pk.h[r] = (_Float16)(acc[sub][r] + bias[ob + r]);
      *(v4u*)(out + (size_t)(n0 + t16) * O + ob) = pk.u;  // b128 store
    } else {
#pragma unroll
      for (int r = 0; r < 8; ++r)
        out[(size_t)(ob + r) * N + n0 + t16] = (_Float16)(acc[sub][r] + bias[ob + r]);
    }
  }
}

// ---------------------------------------------------------------------------
// Flash attention. Grid (N/32, B), block 256 (8 waves).
// WG covers queries [i0, i0+32) x all 512 channels; wave w owns channels
// [w*64, w*64+64). Per 32-key block:
//   stage K tile (32j x 64d f16) in LDS (async-to-LDS if available)
//   S^T(j x i) = K_tile * Q            -> 8 WMMAs (2 i-tiles x 2 j x 2 k)
//   online softmax over j per i-tile
//   O^T(c x i) += V(c x j32) * P^T     -> 8 WMMAs (4 c-subtiles x 2 i-tiles,
//                                         V fragment loaded once per subtile)
// ---------------------------------------------------------------------------
__global__ void __launch_bounds__(256)
attn_kernel(const _Float16* __restrict__ qT,  // [B][N][64]
            const _Float16* __restrict__ kT,  // [B][N][64]
            const _Float16* __restrict__ vF,  // [B][512][N]
            const float* __restrict__ x,      // [B][512][N]
            float* __restrict__ out) {        // [B][512][N]
  const int N = 4096, Cc = 512, D = 64;
  const int KP = 72;                       // padded LDS row stride (halfs)
  __shared__ _Float16 lk[32 * KP];         // K tile: 32 j-rows x 64 d (+pad)

  const int tid  = threadIdx.x;
  const int lane = tid & 31;
  const int wave = tid >> 5;
  const int h = lane >> 4, t16 = lane & 15;
  const int i0 = blockIdx.x * 32;
  const int b  = blockIdx.y;
  const _Float16* qb = qT + (size_t)b * N * D;
  const _Float16* kb = kT + (size_t)b * N * D;
  const _Float16* vb = vF + (size_t)b * Cc * N;
  const int cbase = wave * 64;

  // K-tile staging map: thread t loads 8 halfs of row (t/8), col (t%8)*8.
  const int krow = tid >> 3;
  const int kcol = (tid & 7) * 8;

  // Q fragments (B-matrix): 2 i-tiles x 2 k-steps, loaded once.
  v16h bq[2][2];
#pragma unroll
  for (int it = 0; it < 2; ++it) {
    const _Float16* qr = qb + (size_t)(i0 + it * 16 + t16) * D;
    bq[it][0] = load_b16(qr, 0, h);
    bq[it][1] = load_b16(qr, 32, h);
  }

  float m_run[2] = {-3.0e38f, -3.0e38f};
  float l_run[2] = {0.0f, 0.0f};
  v8f acc[2][4] = {{{}, {}, {}, {}}, {{}, {}, {}, {}}};

  for (int j0 = 0; j0 < N; j0 += 32) {
    __syncthreads();  // protect LDS from previous iteration's readers
#if USE_ASYNC_LDS
    __builtin_amdgcn_global_load_async_to_lds_b128(
        (gv4i_p)(kb + (size_t)(j0 + krow) * D + kcol),
        (lv4i_p)&lk[krow * KP + kcol], 0, 0);
    __builtin_amdgcn_s_wait_asynccnt(0);
#else
    {
      H8 tmp;
      tmp.u = *(const v4u*)(kb + (size_t)(j0 + krow) * D + kcol);
      *(v4u*)&lk[krow * KP + kcol] = tmp.u;
    }
#endif
    __syncthreads();

    // ---- scores S^T: rows j, cols i (per i-tile) -------------------------
    v8f s[2][2] = {{{}, {}}, {{}, {}}};
#pragma unroll
    for (int jt = 0; jt < 2; ++jt) {
      const _Float16* arow = &lk[(jt * 16 + t16) * KP];
      v16h a0 = load_a16(arow, 0, h);   // ds_load_b128 fragments
      v16h a1 = load_a16(arow, 32, h);
#pragma unroll
      for (int it = 0; it < 2; ++it) {
        s[it][jt] = WMMA_F16(a0, bq[it][0], s[it][jt]);
        s[it][jt] = WMMA_F16(a1, bq[it][1], s[it][jt]);
      }
    }

    // ---- online softmax over j, per i-tile; build P^T B-fragments --------
    v16h pb[2];
#pragma unroll
    for (int it = 0; it < 2; ++it) {
      float tmax = s[it][0][0];
#pragma unroll
      for (int r = 1; r < 8; ++r) tmax = fmaxf(tmax, s[it][0][r]);
#pragma unroll
      for (int r = 0; r < 8; ++r) tmax = fmaxf(tmax, s[it][1][r]);
      tmax = fmaxf(tmax, __shfl_xor(tmax, 16, 32));
      const float m_new = fmaxf(m_run[it], tmax);

      float p0[8], p1[8], lsum = 0.0f;
#pragma unroll
      for (int r = 0; r < 8; ++r) { p0[r] = __expf(s[it][0][r] - m_new); lsum += p0[r]; }
#pragma unroll
      for (int r = 0; r < 8; ++r) { p1[r] = __expf(s[it][1][r] - m_new); lsum += p1[r]; }
      lsum += __shfl_xor(lsum, 16, 32);
      const float scale = __expf(m_run[it] - m_new);
      l_run[it] = l_run[it] * scale + lsum;
      m_run[it] = m_new;
#pragma unroll
      for (int sub = 0; sub < 4; ++sub)
#pragma unroll
        for (int r = 0; r < 8; ++r) acc[it][sub][r] *= scale;

      // repack: element t of B-fragment <- P[h*16 + t][i]
#pragma unroll
      for (int r = 0; r < 8; ++r) {
        const float q0 = __shfl_xor(p0[r], 16, 32);
        const float q1 = __shfl_xor(p1[r], 16, 32);
        pb[it][r]     = (_Float16)(h ? q1    : p0[r]);  // j = h*16 + r
        pb[it][8 + r] = (_Float16)(h ? p1[r] : q0);     // j = h*16 + 8 + r
      }
    }

    // ---- O^T += V * P^T (V fragment shared by both i-tiles) --------------
#pragma unroll
    for (int sub = 0; sub < 4; ++sub) {
      const _Float16* vr = vb + (size_t)(cbase + sub * 16 + t16) * N + j0;
      v16h av = load_a16(vr, 0, h);
      acc[0][sub] = WMMA_F16(av, pb[0], acc[0][sub]);
      acc[1][sub] = WMMA_F16(av, pb[1], acc[1][sub]);
    }
  }

  // ---- epilogue: normalize + residual -----------------------------------
  const float* xb = x   + (size_t)b * Cc * N;
  float*       ob = out + (size_t)b * Cc * N;
#pragma unroll
  for (int it = 0; it < 2; ++it) {
    const float inv_l = 1.0f / l_run[it];
    const size_t ii = (size_t)(i0 + it * 16 + t16);
#pragma unroll
    for (int sub = 0; sub < 4; ++sub) {
#pragma unroll
      for (int r = 0; r < 8; ++r) {
        const size_t c = (size_t)(cbase + sub * 16 + 8 * h + r);
        ob[c * N + ii] = acc[it][sub][r] * inv_l + xb[c * N + ii];
      }
    }
  }
}

// ---------------------------------------------------------------------------
extern "C" void kernel_launch(void* const* d_in, const int* in_sizes, int n_in,
                              void* d_out, int out_size, void* d_ws, size_t ws_size,
                              hipStream_t stream) {
  (void)in_sizes; (void)n_in; (void)out_size; (void)ws_size;
  const float* x  = (const float*)d_in[0];
  const float* Wq = (const float*)d_in[1];
  const float* bq = (const float*)d_in[2];
  const float* Wk = (const float*)d_in[3];
  const float* bk = (const float*)d_in[4];
  const float* Wv = (const float*)d_in[5];
  const float* bv = (const float*)d_in[6];
  float* out = (float*)d_out;

  const int B = 4, Cc = 512, N = 4096, D = 64;

  _Float16* qT = (_Float16*)d_ws;                 // [B][N][64]   f16
  _Float16* kT = qT + (size_t)B * N * D;          // [B][N][64]   f16
  _Float16* vF = kT + (size_t)B * N * D;          // [B][512][N]  f16

  dim3 gq(N / 128, D / 64, B);    // (32, 1, 4)
  proj64_kernel<<<gq, 256, 0, stream>>>(x, Wq, bq, qT, D, 1);
  proj64_kernel<<<gq, 256, 0, stream>>>(x, Wk, bk, kT, D, 1);
  dim3 gv(N / 128, Cc / 64, B);   // (32, 8, 4)
  proj64_kernel<<<gv, 256, 0, stream>>>(x, Wv, bv, vF, Cc, 0);

  dim3 ga(N / 32, B);             // (128, 4)
  attn_kernel<<<ga, 256, 0, stream>>>(qT, kT, vF, x, out);
}
